// LVM_3487513444979
// MI455X (gfx1250) — matrix operand order repaired and checked by reading, not compiled, not run on hardware
//
#include <hip/hip_runtime.h>
#include <hip/hip_bf16.h>
#include <math.h>

typedef __bf16 bf16;
typedef __attribute__((ext_vector_type(16))) __bf16 v16bf;
typedef __attribute__((ext_vector_type(8)))  __bf16 v8bf;
typedef __attribute__((ext_vector_type(8)))  float  v8f;

// ---------------- problem constants ----------------
constexpr int NPIX = 1800;   // B * 30 * 30
constexpr int NPAD = 1824;   // padded row stride (activations laid out [C][NPAD])
constexpr int NT   = 29;     // ceil(1800/64) N-tiles
constexpr int EMB  = 256;
constexpr int NH   = 4;
constexpr int DH   = 64;
constexpr int DFF  = 2048;
constexpr int NCLS = 16;
constexpr int MAXL = 16;
constexpr int BATCH= 2;
constexpr int SA   = 40;     // LDS row stride (bf16): 80B rows stay 16B aligned

// ---------------- fp32 -> bf16 weight panel, zero-padded to KP ----------------
__global__ __launch_bounds__(256)
void cvt_w_k(const float* __restrict__ W, bf16* __restrict__ dst, int K, int KP)
{
  int r = blockIdx.x;
  for (int k = threadIdx.x; k < KP; k += 256)
    dst[(size_t)r * KP + k] = (k < K) ? (bf16)W[(size_t)r * K + k] : (bf16)0.0f;
}

// ============================================================
// Implicit-GEMM conv (3x3 SAME or 1x1) via v_wmma_f32_16x16x32_bf16
//   act : bf16 [Cin][NPAD], n = b*900 + y*30 + x
//   Wbf : bf16 [Cout][KP]   (pre-converted, zero padded, KP = K round-up 32)
//   out : fp32 [Cout][NPAD], out = conv + bias (pre-BN)
// 128 threads (4 wave32), 64x64 tile, double-buffered LDS.
// A panel streamed with GLOBAL_LOAD_ASYNC_TO_LDS_B128 (ASYNCcnt),
// overlapped with im2col gather and the WMMA pipe.
// ============================================================
__global__ __launch_bounds__(128)
void conv_gemm_bf16(const bf16* __restrict__ act,
                    const bf16* __restrict__ Wbf,
                    const float* __restrict__ bias,
                    float* __restrict__ out,
                    int Cin, int KK, int KP)
{
  const int K    = Cin * KK;            // real reduction size (im2col predicate)
  const int tm   = blockIdx.x;          // Cout/64 tiles
  const int tn   = blockIdx.y;          // 29 N tiles
  const int tid  = threadIdx.x;
  const int wave = tid >> 5;
  const int lane = tid & 31;
  const int l16  = lane & 15;
  const int hf   = lane >> 4;           // half-wave select

  __shared__ __attribute__((aligned(16))) bf16 As[2][64 * SA];   // [m][k]
  __shared__ __attribute__((aligned(16))) bf16 Bs[2][64 * SA];   // [n][k] (transposed im2col)

  const uint32_t asb[2] = { (uint32_t)(uintptr_t)&As[0][0],
                            (uint32_t)(uintptr_t)&As[1][0] };

  v8f acc[4];
#pragma unroll
  for (int j = 0; j < 4; ++j)
#pragma unroll
    for (int i = 0; i < 8; ++i) acc[j][i] = 0.0f;

  // ---- async DMA of a 64x32 bf16 weight panel straight into LDS ----
  auto issue_a = [&](int buf, int k0) {
#pragma unroll
    for (int r = 0; r < 2; ++r) {
      int c = tid + r * 128;            // 256 chunks of 16B
      int m = c >> 2, q = c & 3;
      uint32_t lds = asb[buf] + (uint32_t)(m * (SA * 2) + q * 16);
      uint64_t ga  = (uint64_t)(uintptr_t)(Wbf + (size_t)(tm * 64 + m) * KP + (k0 + q * 8));
      asm volatile("global_load_async_to_lds_b128 %0, %1, off"
                   :: "v"(lds), "v"(ga) : "memory");
    }
  };

  // ---- im2col gather, stored transposed [n][k] (coalesced over n) ----
  auto stage_b = [&](int buf, int k0) {
#pragma unroll
    for (int i = 0; i < 16; ++i) {
      int e  = i * 128 + tid;
      int kk = e >> 6;                  // 0..31
      int nn = e & 63;                  // 0..63
      int kG = k0 + kk;
      int nG = tn * 64 + nn;
      bf16 v = (bf16)0.0f;
      if (kG < K && nG < NPIX) {
        int b = nG / 900;
        int p = nG - b * 900;
        int y = p / 30;
        int x = p - y * 30;
        int ci;
        if (KK == 9) {
          ci = kG / 9;
          int r = kG - ci * 9;
          y += r / 3 - 1;
          x += r - (r / 3) * 3 - 1;
        } else {
          ci = kG;
        }
        if ((unsigned)y < 30u && (unsigned)x < 30u)
          v = act[(size_t)ci * NPAD + b * 900 + y * 30 + x];
      }
      Bs[buf][nn * SA + kk] = v;
    }
  };

  // ---- fragment loads + 4 WMMAs on one buffer ----
  auto compute = [&](int buf) {
    // A: 16-bit layout, half 0 -> K {0..7,16..23}, half 1 -> K {8..15,24..31}
    v16bf afr;
    {
      const bf16* ar = &As[buf][(wave * 16 + l16) * SA + hf * 8];
      v8bf a0 = *(const v8bf*)(ar);
      v8bf a1 = *(const v8bf*)(ar + 16);
#pragma unroll
      for (int i = 0; i < 8; ++i) { afr[i] = a0[i]; afr[8 + i] = a1[i]; }
    }
#pragma unroll
    for (int j = 0; j < 4; ++j) {
      // B: same as A, half 0 -> K {0..7,16..23}, half 1 -> K {8..15,24..31}
      const bf16* br = &Bs[buf][(j * 16 + l16) * SA + hf * 8];
      v8bf b0 = *(const v8bf*)(br);
      v8bf b1 = *(const v8bf*)(br + 16);
      v16bf bfr;
#pragma unroll
      for (int i = 0; i < 8; ++i) { bfr[i] = b0[i]; bfr[8 + i] = b1[i]; }
      acc[j] = __builtin_amdgcn_wmma_f32_16x16x32_bf16(
          false, afr, false, bfr, (short)0, acc[j], false, false);
    }
  };

  // ---- software pipeline: prologue ----
  issue_a(0, 0);
  stage_b(0, 0);
  asm volatile("s_wait_asynccnt 0" ::: "memory");
  __syncthreads();

  int it = 0;
  for (int k0 = 0; k0 < KP; k0 += 32, ++it) {
    int cur = it & 1;
    bool more = (k0 + 32) < KP;
    if (more) {
      issue_a(cur ^ 1, k0 + 32);        // DMA next weight panel (ASYNCcnt)
      stage_b(cur ^ 1, k0 + 32);        // gather next im2col panel
    }
    compute(cur);                        // WMMA on current buffers
    if (more) asm volatile("s_wait_asynccnt 0" ::: "memory");
    __syncthreads();
  }

  // ---- epilogue: C layout VGPR i -> M = i + 8*half, N = l16 ----
#pragma unroll
  for (int j = 0; j < 4; ++j) {
    int nG = tn * 64 + j * 16 + l16;
    if (nG >= NPIX) continue;
#pragma unroll
    for (int i = 0; i < 8; ++i) {
      int mG = tm * 64 + wave * 16 + hf * 8 + i;
      out[(size_t)mG * NPAD + nG] = acc[j][i] + bias[mG];
    }
  }
}

// ---------------- BatchNorm statistics -> scale/shift ----------------
__global__ __launch_bounds__(256)
void bn_stats_k(const float* __restrict__ x, const float* __restrict__ g,
                const float* __restrict__ be, float* __restrict__ scale,
                float* __restrict__ shift)
{
  int c = blockIdx.x;
  const float* row = x + (size_t)c * NPAD;
  float s = 0.f, s2 = 0.f;
  for (int n = threadIdx.x; n < NPIX; n += 256) {
    float v = row[n];
    s += v; s2 += v * v;
  }
  __shared__ float ls[256], lq[256];
  ls[threadIdx.x] = s; lq[threadIdx.x] = s2;
  __syncthreads();
  for (int off = 128; off > 0; off >>= 1) {
    if (threadIdx.x < off) { ls[threadIdx.x] += ls[threadIdx.x + off]; lq[threadIdx.x] += lq[threadIdx.x + off]; }
    __syncthreads();
  }
  if (threadIdx.x == 0) {
    float mean = ls[0] * (1.0f / NPIX);
    float var  = lq[0] * (1.0f / NPIX) - mean * mean;
    float sc   = g[c] * rsqrtf(var + 1e-5f);
    scale[c] = sc;
    shift[c] = be[c] - mean * sc;
  }
}

// ---------------- BN apply + ReLU + fp32 -> bf16 ----------------
__global__ __launch_bounds__(256)
void bn_apply_k(const float* __restrict__ x, const float* __restrict__ scale,
                const float* __restrict__ shift, bf16* __restrict__ dst)
{
  int c = blockIdx.x;
  float sc = scale[c], sh = shift[c];
  const float* src = x + (size_t)c * NPAD;
  bf16* d = dst + (size_t)c * NPAD;
  for (int n = threadIdx.x; n < NPIX; n += 256)
    d[n] = (bf16)fmaxf(src[n] * sc + sh, 0.0f);
}

// ---------------- input NCHW fp32 -> [C][NPAD] bf16 ----------------
__global__ void pack_input_k(const float* __restrict__ x, bf16* __restrict__ dst)
{
  int c = blockIdx.x;                    // 10
  for (int n = threadIdx.x; n < NPIX; n += blockDim.x) {
    int b = n / 900, p = n - b * 900;
    dst[(size_t)c * NPAD + n] = (bf16)x[(size_t)(b * 10 + c) * 900 + p];
  }
}

// ---------------- projection: mem[b][e] = flat[b] . proj_W[e] + bias ----
// 472 MB fp32 weight stream -> HBM-bound; prefetch the stream.
__global__ __launch_bounds__(256)
void proj_k(const float* __restrict__ F,      // [512][NPAD]
            const float* __restrict__ Wp,     // [256][460800]
            const float* __restrict__ bp,
            float* __restrict__ memo)         // [2][256]
{
  int e = blockIdx.x;
  const float* w = Wp + (size_t)e * 460800;
  float s0 = 0.f, s1 = 0.f;
  for (int k = threadIdx.x; k < 460800; k += 256) {
    if (k + 4096 < 460800) __builtin_prefetch(w + k + 4096, 0, 1);
    int c = k / 900, p = k - c * 900;
    float wv = w[k];
    const float* fr = F + (size_t)c * NPAD + p;
    s0 += wv * fr[0];
    s1 += wv * fr[900];
  }
  __shared__ float r0[256], r1[256];
  r0[threadIdx.x] = s0; r1[threadIdx.x] = s1;
  __syncthreads();
  for (int off = 128; off > 0; off >>= 1) {
    if (threadIdx.x < off) { r0[threadIdx.x] += r0[threadIdx.x + off]; r1[threadIdx.x] += r1[threadIdx.x + off]; }
    __syncthreads();
  }
  if (threadIdx.x == 0) {
    memo[e]       = r0[0] + bp[e];
    memo[EMB + e] = r1[0] + bp[e];
  }
}

// ---------------- tiny row-major GEMM: out[r][m] = in[r].W[m] + b ----
__global__ void sgemm_k(const float* __restrict__ in, const float* __restrict__ W,
                        const float* __restrict__ bias, float* __restrict__ out,
                        int R, int M, int Kd, int relu)
{
  int idx = blockIdx.x * blockDim.x + threadIdx.x;
  if (idx >= R * M) return;
  int r = idx / M, m = idx - r * M;
  const float* a = in + (size_t)r * Kd;
  const float* w = W + (size_t)m * Kd;
  float s = bias[m];
  for (int k = 0; k < Kd; ++k) s += a[k] * w[k];
  if (relu) s = fmaxf(s, 0.0f);
  out[idx] = s;
}

// ---------------- self-attention (rows = l*B+b, heads = 4, dh = 64) ----
__global__ __launch_bounds__(64)
void sa_attn_k(const float* __restrict__ qkv, float* __restrict__ o, int L, int B)
{
  int blk = blockIdx.x;
  int h  = blk % NH;
  int lb = blk / NH;
  int b  = lb % B;
  int l  = lb / B;
  int j  = threadIdx.x;                 // 0..63

  __shared__ float red[DH];
  __shared__ float att[MAXL];

  float q = qkv[(size_t)(l * B + b) * 3 * EMB + h * DH + j];
  for (int s = 0; s < L; ++s) {
    red[j] = q * qkv[(size_t)(s * B + b) * 3 * EMB + EMB + h * DH + j];
    __syncthreads();
    for (int off = 32; off > 0; off >>= 1) {
      if (j < off) red[j] += red[j + off];
      __syncthreads();
    }
    if (j == 0) att[s] = red[0] * 0.125f;   // 1/sqrt(64)
    __syncthreads();
  }
  if (j == 0) {
    float mx = att[0];
    for (int s = 1; s < L; ++s) mx = fmaxf(mx, att[s]);
    float sum = 0.f;
    for (int s = 0; s < L; ++s) { att[s] = __expf(att[s] - mx); sum += att[s]; }
    float inv = 1.0f / sum;
    for (int s = 0; s < L; ++s) att[s] *= inv;
  }
  __syncthreads();
  float accv = 0.f;
  for (int s = 0; s < L; ++s)
    accv += att[s] * qkv[(size_t)(s * B + b) * 3 * EMB + 2 * EMB + h * DH + j];
  o[(size_t)(l * B + b) * EMB + h * DH + j] = accv;
}

// ---------------- residual add + LayerNorm over EMB=256 ----------------
__global__ __launch_bounds__(EMB)
void add_ln_k(const float* __restrict__ x, const float* __restrict__ y,
              const float* __restrict__ g, const float* __restrict__ be,
              float* __restrict__ dst, int ymod, int B)
{
  int r = blockIdx.x;
  int e = threadIdx.x;
  int yr = ymod ? (r % B) : r;
  float v = x[(size_t)r * EMB + e] + y[(size_t)yr * EMB + e];
  __shared__ float red[EMB];
  red[e] = v;
  __syncthreads();
  for (int off = EMB / 2; off > 0; off >>= 1) {
    if (e < off) red[e] += red[e + off];
    __syncthreads();
  }
  float mean = red[0] * (1.0f / EMB);
  __syncthreads();
  float d = v - mean;
  red[e] = d * d;
  __syncthreads();
  for (int off = EMB / 2; off > 0; off >>= 1) {
    if (e < off) red[e] += red[e + off];
    __syncthreads();
  }
  float var = red[0] * (1.0f / EMB);
  dst[(size_t)r * EMB + e] = d * rsqrtf(var + 1e-5f) * g[e] + be[e];
}

// ---------------- token embedding + positional encoding ----------------
__global__ void embed_k(const int* __restrict__ tok, const float* __restrict__ table,
                        const float* __restrict__ pe, float* __restrict__ h0, int B)
{
  int r = blockIdx.x;            // l*B + b
  int e = threadIdx.x;
  int l = r / B, b = r - l * B;
  int t = tok[b * (MAXL + 1) + l];
  h0[(size_t)r * EMB + e] = table[(size_t)t * EMB + e] + pe[(size_t)l * EMB + e];
}

__global__ void pe_init_k(float* __restrict__ pe)
{
  int p = blockIdx.x, e = threadIdx.x;
  float f = __expf(-(float)((e >> 1) * 2) * (9.210340371976184f / 256.0f));
  float a = (float)p * f;
  pe[(size_t)p * EMB + e] = (e & 1) ? cosf(a) : sinf(a);
}

__global__ void tok_init_k(int* __restrict__ tok)
{
  if (threadIdx.x < BATCH) tok[threadIdx.x * (MAXL + 1)] = 3;   // START_TOK
}

// ---------------- output head: logits, write d_out, argmax next tok ----
__global__ void head_k(const float* __restrict__ h, const float* __restrict__ Wo,
                       const float* __restrict__ bo, int* __restrict__ tok,
                       float* __restrict__ out, int L, int B, int step)
{
  int b = blockIdx.x;
  int c = threadIdx.x;              // 0..15
  const float* row = h + (size_t)((L - 1) * B + b) * EMB;
  float s = bo[c];
  for (int k = 0; k < EMB; ++k) s += row[k] * Wo[(size_t)c * EMB + k];
  __shared__ float lg[NCLS];
  lg[c] = s;
  out[(size_t)(b * MAXL + step) * NCLS + c] = s;
  __syncthreads();
  if (c == 0) {
    int best = 0; float bv = lg[0];
    for (int i = 1; i < NCLS; ++i) if (lg[i] > bv) { bv = lg[i]; best = i; }
    tok[b * (MAXL + 1) + step + 1] = best;
  }
}

// ============================================================
// Host orchestration
// ============================================================
extern "C" void kernel_launch(void* const* d_in, const int* in_sizes, int n_in,
                              void* d_out, int out_size, void* d_ws, size_t ws_size,
                              hipStream_t stream)
{
  (void)in_sizes; (void)n_in; (void)out_size;
  const float* x1 = (const float*)d_in[0];
  const float* x2 = (const float*)d_in[1];
  auto P = [&](int i) { return (const float*)d_in[i]; };

  // ---- workspace bump allocator ----
  char* base = (char*)d_ws;
  size_t off = 0;
  auto alloc = [&](size_t bytes) -> char* {
    char* p = base + off;
    off += (bytes + 255) & ~(size_t)255;
    return p;
  };
  float* convf  = (float*)alloc((size_t)2048 * NPAD * 4);
  float* finalf = (float*)alloc((size_t)512  * NPAD * 4);
  bf16*  wbf    = (bf16*) alloc((size_t)2048 * 18432 * 2);  // largest bf16 weight panel (bridge)
  bf16*  tmpA   = (bf16*) alloc((size_t)2048 * NPAD * 2);
  bf16*  tmpB   = (bf16*) alloc((size_t)2048 * NPAD * 2);
  bf16*  xin    = (bf16*) alloc((size_t)16   * NPAD * 2);
  bf16*  catB   = (bf16*) alloc((size_t)2048 * NPAD * 2);   // f14 | f24
  bf16*  cat3   = (bf16*) alloc((size_t)3072 * NPAD * 2);   // bridge | f13 | f23
  bf16*  cat2   = (bf16*) alloc((size_t)1536 * NPAD * 2);   // up3 | f12 | f22
  bf16*  cat1   = (bf16*) alloc((size_t)768  * NPAD * 2);   // up2 | f11 | f21
  float* scale  = (float*)alloc(2048 * 4);
  float* shift  = (float*)alloc(2048 * 4);
  float* memb   = (float*)alloc(BATCH * EMB * 4);
  float* pe     = (float*)alloc(MAXL * EMB * 4);
  int*   tok    = (int*)  alloc(BATCH * (MAXL + 1) * 4);
  float* h0     = (float*)alloc(32 * EMB * 4);
  float* qkv    = (float*)alloc(32 * 3 * EMB * 4);
  float* sao    = (float*)alloc(32 * EMB * 4);
  float* saout  = (float*)alloc(32 * EMB * 4);
  float* xb     = (float*)alloc(32 * EMB * 4);
  float* x2b    = (float*)alloc(32 * EMB * 4);
  float* ffh    = (float*)alloc(32 * DFF * 4);
  float* ffo    = (float*)alloc(32 * EMB * 4);
  float* hb     = (float*)alloc(32 * EMB * 4);
  float* catmp  = (float*)alloc(BATCH * EMB * 4);
  float* caadd  = (float*)alloc(BATCH * EMB * 4);
  if (off > ws_size) return;

  // conv + BN + ReLU (bf16 weights pre-converted; dst may be a concat slot)
  auto cbr = [&](const bf16* in, int Cin, int KK, int pw, int Cout, bf16* dst) {
    int K  = Cin * KK;
    int KP = (K + 31) & ~31;
    cvt_w_k<<<Cout, 256, 0, stream>>>(P(pw), wbf, K, KP);
    conv_gemm_bf16<<<dim3(Cout / 64, NT), 128, 0, stream>>>(in, wbf, P(pw + 1), convf, Cin, KK, KP);
    bn_stats_k<<<Cout, 256, 0, stream>>>(convf, P(pw + 2), P(pw + 3), scale, shift);
    bn_apply_k<<<Cout, 256, 0, stream>>>(convf, scale, shift, dst);
  };
  auto block3 = [&](const bf16* in, int Cin, int Cout, int pbase, bf16* dst) {
    cbr(in,   Cin,  9, pbase + 0, Cout, tmpA);
    cbr(tmpA, Cout, 9, pbase + 4, Cout, tmpB);
    cbr(tmpB, Cout, 9, pbase + 8, Cout, dst);
  };

  // ---- dual encoders, writing skip features directly into concat buffers ----
  auto encoder = [&](const float* x, int e) {
    pack_input_k<<<10, 256, 0, stream>>>(x, xin);
    bf16* f1 = cat1 + (size_t)(512  + e * 128) * NPAD;
    bf16* f2 = cat2 + (size_t)(1024 + e * 256) * NPAD;
    bf16* f3 = cat3 + (size_t)(2048 + e * 512) * NPAD;
    bf16* f4 = catB + (size_t)(e * 1024) * NPAD;
    block3(xin, 10,  128,  2, f1);
    block3(f1, 128,  256, 14, f2);
    block3(f2, 256,  512, 26, f3);
    block3(f3, 512, 1024, 38, f4);
  };
  encoder(x1, 0);
  encoder(x2, 1);

  block3(catB, 2048, 2048, 50, cat3);   // bridge -> cat3[0:2048]
  block3(cat3, 3072, 1024, 62, cat2);   // up3    -> cat2[0:1024]
  block3(cat2, 1536,  512, 74, cat1);   // up2    -> cat1[0:512]
  block3(cat1,  768,  512, 86, tmpA);   // up1    -> tmpA

  // final 1x1 conv (bias only, no BN/ReLU)
  cvt_w_k<<<512, 256, 0, stream>>>(P(98), wbf, 512, 512);
  conv_gemm_bf16<<<dim3(512 / 64, NT), 128, 0, stream>>>(tmpA, wbf, P(99), finalf, 512, 1, 512);

  // projection to memory tokens
  proj_k<<<EMB, 256, 0, stream>>>(finalf, P(100), P(101), memb);

  // cross-attention over S=1 memory: softmax == 1 -> constant per-batch add
  sgemm_k<<<dim3((BATCH * EMB + 255) / 256), 256, 0, stream>>>(
      memb, P(107) + (size_t)2 * EMB * EMB, P(108) + 2 * EMB, catmp, BATCH, EMB, EMB, 0);
  sgemm_k<<<dim3((BATCH * EMB + 255) / 256), 256, 0, stream>>>(
      catmp, P(109), P(110), caadd, BATCH, EMB, EMB, 0);

  pe_init_k<<<MAXL, EMB, 0, stream>>>(pe);
  tok_init_k<<<1, 32, 0, stream>>>(tok);

  // ---- autoregressive decode: full re-decode each step (faithful) ----
  for (int step = 0; step < MAXL; ++step) {
    int L = step + 1;
    int R = L * BATCH;
    embed_k<<<R, EMB, 0, stream>>>(tok, P(102), pe, h0, BATCH);
    sgemm_k<<<dim3((R * 3 * EMB + 255) / 256), 256, 0, stream>>>(h0, P(103), P(104), qkv, R, 3 * EMB, EMB, 0);
    sa_attn_k<<<R * NH, DH, 0, stream>>>(qkv, sao, L, BATCH);
    sgemm_k<<<dim3((R * EMB + 255) / 256), 256, 0, stream>>>(sao, P(105), P(106), saout, R, EMB, EMB, 0);
    add_ln_k<<<R, EMB, 0, stream>>>(h0, saout, P(115), P(116), xb, 0, BATCH);
    add_ln_k<<<R, EMB, 0, stream>>>(xb, caadd, P(117), P(118), x2b, 1, BATCH);
    sgemm_k<<<dim3((R * DFF + 255) / 256), 256, 0, stream>>>(x2b, P(111), P(112), ffh, R, DFF, EMB, 1);
    sgemm_k<<<dim3((R * EMB + 255) / 256), 256, 0, stream>>>(ffh, P(113), P(114), ffo, R, EMB, DFF, 0);
    add_ln_k<<<R, EMB, 0, stream>>>(x2b, ffo, P(119), P(120), hb, 0, BATCH);
    head_k<<<BATCH, NCLS, 0, stream>>>(hb, P(121), P(122), tok, (float*)d_out, L, BATCH, step);
  }
}
